// SimpleRNN_13915694039230
// MI455X (gfx1250) — matrix operand order repaired and checked by reading, not compiled, not run on hardware
//
#include <hip/hip_runtime.h>
#include <math.h>

typedef __attribute__((ext_vector_type(16))) _Float16 v16h;
typedef __attribute__((ext_vector_type(2)))  _Float16 h2;
typedef __attribute__((ext_vector_type(8)))  float    v8f;
typedef __attribute__((ext_vector_type(8)))  int      v8i;

#define B_ 4096
#define S_ 512
#define I_ 28
#define H_ 128
#define O_ 28

#if __has_builtin(__builtin_amdgcn_tanhf)
  #define TANHF(v) __builtin_amdgcn_tanhf(v)
#elif __has_builtin(__builtin_amdgcn_tanh_f32)
  #define TANHF(v) __builtin_amdgcn_tanh_f32(v)
#else
  #define TANHF(v) tanhf(v)
#endif

// D = A*B + C, f16 inputs, f32 accumulate (v_wmma_f32_16x16x32_f16)
#define WMMA(a, b, c) \
  __builtin_amdgcn_wmma_f32_16x16x32_f16(false, (a), false, (b), (short)0, (c), false, false)

// swap lanes l <-> l^16 on a packed b32 (ds_swizzle group-of-32: xor=0x10, and=0x1f)
__device__ __forceinline__ int swz16(int v) {
  return __builtin_amdgcn_ds_swizzle(v, 0x401f);
}

// pack two f32 -> one b32 holding 2x f16 (v_cvt_pk_rtz_f16_f32 when available)
__device__ __forceinline__ int pk2(float a, float b) {
#if __has_builtin(__builtin_amdgcn_cvt_pkrtz)
  return __builtin_bit_cast(int, __builtin_amdgcn_cvt_pkrtz(a, b));
#else
  h2 v; v[0] = (_Float16)a; v[1] = (_Float16)b;
  return __builtin_bit_cast(int, v);
#endif
}

// A-fragment element -> K index (contiguous, matching the B fills):
// lanes 0-15 hold K 0..15; lanes 16-31 hold K 16..31
__device__ __forceinline__ int a_k(int half, int j) {
  return (half ? 16 : 0) + j;
}

__global__ __launch_bounds__(64, 1) void rnn_fused_wmma(
    const float* __restrict__ x,    const float* __restrict__ W_ih,
    const float* __restrict__ b_ih, const float* __restrict__ W_hh,
    const float* __restrict__ b_hh, const float* __restrict__ W_ho,
    const float* __restrict__ b_ho, float* __restrict__ out)
{
  const int lane = threadIdx.x & 31;
  const int wave = threadIdx.x >> 5;
  const int half = lane >> 4;     // which 16-lane half
  const int m    = lane & 15;     // A row / C column within tile
  const int b0   = (blockIdx.x * 2 + wave) * 16;  // 16 batch rows per wave
  const int row  = b0 + m;        // this lane's batch row (as B/C column)

  // ---------------- register-resident weight fragments (A layout = W^T) -----
  // A_hh[to][c]: 16x32 tile, M = hout in [16*to,16*to+16), K = hin in [32c,32c+32)
  v16h A_hh[8][4];
  #pragma unroll
  for (int to = 0; to < 8; ++to)
    #pragma unroll
    for (int c = 0; c < 4; ++c)
      #pragma unroll
      for (int j = 0; j < 16; ++j) {
        int K = 32 * c + a_k(half, j);
        A_hh[to][c][j] = (_Float16)W_hh[K * H_ + 16 * to + m];
      }

  // A_ih[to]: M = hout, K = input dim (28, zero-padded to 32)
  v16h A_ih[8];
  #pragma unroll
  for (int to = 0; to < 8; ++to)
    #pragma unroll
    for (int j = 0; j < 16; ++j) {
      int K = a_k(half, j);
      float v = (K < I_) ? W_ih[K * H_ + 16 * to + m] : 0.0f;
      A_ih[to][j] = (_Float16)v;
    }

  // combined bias (b_ih + b_hh) in C layout: element r -> hout = 16to + r + 8*half
  v8f biasC[8];
  #pragma unroll
  for (int to = 0; to < 8; ++to)
    #pragma unroll
    for (int r = 0; r < 8; ++r) {
      int h = 16 * to + r + 8 * half;
      biasC[to][r] = b_ih[h] + b_hh[h];
    }

  // ---------------- x^T B-fragment loader (K = input dim, N = batch) --------
  // B 32x16 layout: lanes 0-15 hold K=0..15, lanes 16-31 hold K=16..31
  const float4* xr = (const float4*)(x + (size_t)row * ((size_t)S_ * I_));
  const int qbase = half ? 4 : 0;
  auto load_x = [&](int t) -> v8i {
    const float4* p = xr + (size_t)t * (I_ / 4);
    float4 f0 = p[qbase + 0];
    float4 f1 = p[qbase + 1];
    float4 f2 = p[qbase + 2];
    float4 f3 = p[half ? 6 : 3];          // upper half: dummy (zeroed below)
    float vv[16] = {f0.x, f0.y, f0.z, f0.w, f1.x, f1.y, f1.z, f1.w,
                    f2.x, f2.y, f2.z, f2.w, f3.x, f3.y, f3.z, f3.w};
    v8i r;
    #pragma unroll
    for (int p2 = 0; p2 < 8; ++p2) {
      float a = vv[2 * p2], b = vv[2 * p2 + 1];
      if (2 * p2 >= 12) { a = half ? 0.0f : a; b = half ? 0.0f : b; } // pad i>=28
      r[p2] = pk2(a, b);
    }
    return r;
  };

  // ---------------- recurrence over time ------------------------------------
  v8i bh[4];                                   // h^T B-fragments, f16-packed (h0 = 0)
  #pragma unroll
  for (int c = 0; c < 4; ++c)
    #pragma unroll
    for (int p = 0; p < 8; ++p) bh[c][p] = 0;

  v8i bx = load_x(0);

  #pragma unroll 1
  for (int t = 0; t < S_; ++t) {
    // prefetch + software-pipelined load of next timestep's x
    if (t + 8 < S_) __builtin_prefetch(xr + (size_t)(t + 8) * (I_ / 4), 0, 0);
    const int tn = (t + 1 < S_) ? (t + 1) : (S_ - 1);
    v8i bxn = load_x(tn);

    // x_t^T projection seeds the accumulators with C = bias (no copy movs)
    v8f acc[8];
    #pragma unroll
    for (int to = 0; to < 8; ++to)
      acc[to] = WMMA(A_ih[to], __builtin_bit_cast(v16h, bx), biasC[to]);

    // W_hh^T * h^T : 4 K-chunks x 8 H-tiles (independent accs 8 apart)
    #pragma unroll
    for (int c = 0; c < 4; ++c)
      #pragma unroll
      for (int to = 0; to < 8; ++to)
        acc[to] = WMMA(A_hh[to][c], __builtin_bit_cast(v16h, bh[c]), acc[to]);

    // tanh (v_tanh_f32 transcendental, co-executes with XDL)
    float th[8][8];
    #pragma unroll
    for (int to = 0; to < 8; ++to)
      #pragma unroll
      for (int r = 0; r < 8; ++r) th[to][r] = TANHF(acc[to][r]);

    // Re-layout C -> B fragments for next step, on packed f16 pairs:
    // per pair: 2 packs, 1 swizzle each, 1 cndmask_b32 each direction.
    // B elem j (j<8): lanes<16 take tile 2c same-lane, lanes>=16 take tile 2c+1
    // from lane^16 (and symmetrically for j>=8).
    #pragma unroll
    for (int c = 0; c < 4; ++c)
      #pragma unroll
      for (int p = 0; p < 4; ++p) {          // element pair j = 2p, 2p+1
        int pa  = pk2(th[2 * c][2 * p],     th[2 * c][2 * p + 1]);
        int pb  = pk2(th[2 * c + 1][2 * p], th[2 * c + 1][2 * p + 1]);
        int pas = swz16(pa);
        int pbs = swz16(pb);
        bh[c][p]     = half ? pbs : pa;      // elements j, j+1
        bh[c][p + 4] = half ? pb  : pas;     // elements j+8, j+9
      }

    bx = bxn;
  }

  // ---------------- output projection: out^T = W_ho^T * h_final^T -----------
  v16h A_ho[2][4];
  #pragma unroll
  for (int to = 0; to < 2; ++to)
    #pragma unroll
    for (int c = 0; c < 4; ++c)
      #pragma unroll
      for (int j = 0; j < 16; ++j) {
        int K = 32 * c + a_k(half, j);
        int o = 16 * to + m;
        float v = (o < O_) ? W_ho[K * O_ + o] : 0.0f;
        A_ho[to][c][j] = (_Float16)v;
      }

  v8f oacc[2];
  #pragma unroll
  for (int to = 0; to < 2; ++to)
    #pragma unroll
    for (int r = 0; r < 8; ++r) oacc[to][r] = 0.0f;

  #pragma unroll
  for (int c = 0; c < 4; ++c)
    #pragma unroll
    for (int to = 0; to < 2; ++to)
      oacc[to] = WMMA(A_ho[to][c], __builtin_bit_cast(v16h, bh[c]), oacc[to]);

  #pragma unroll
  for (int to = 0; to < 2; ++to)
    #pragma unroll
    for (int r = 0; r < 8; ++r) {
      int o = 16 * to + r + 8 * half;
      if (o < O_) out[(size_t)row * O_ + o] = oacc[to][r] + b_ho[o];
    }
}

extern "C" void kernel_launch(void* const* d_in, const int* in_sizes, int n_in,
                              void* d_out, int out_size, void* d_ws, size_t ws_size,
                              hipStream_t stream) {
  const float* x    = (const float*)d_in[0];
  const float* W_ih = (const float*)d_in[1];
  const float* b_ih = (const float*)d_in[2];
  const float* W_hh = (const float*)d_in[3];
  const float* b_hh = (const float*)d_in[4];
  const float* W_ho = (const float*)d_in[5];
  const float* b_ho = (const float*)d_in[6];
  float* out = (float*)d_out;

  dim3 grid(B_ / 32);   // 128 blocks x 2 waves x 16 batch rows = 4096
  dim3 block(64);
  rnn_fused_wmma<<<grid, block, 0, stream>>>(x, W_ih, b_ih, W_hh, b_hh,
                                             W_ho, b_ho, out);
}
